// NearestNeighborGraph_40209483825557
// MI455X (gfx1250) — matrix operand order repaired and checked by reading, not compiled, not run on hardware
//
#include <hip/hip_runtime.h>

typedef __attribute__((ext_vector_type(2)))  float  v2f;
typedef __attribute__((ext_vector_type(8)))  float  v8f;
typedef __attribute__((ext_vector_type(16))) __bf16 v16bf;
typedef unsigned short ushort_t;

#define BB 64
#define LL 1024
#define DD 256
#define KK 16

// ---------------- fp32 fallback LDS layout (floats) ----------------
#define A_STRIDE 260
#define D_STRIDE 1044
#define SQ_OFF   (16 * A_STRIDE)
#define DIST_OFF (SQ_OFF + LL)
#define LDS_F32_FLOATS (DIST_OFF + 16 * D_STRIDE)          /* 21888 -> 87552 B */

// ---------------- bf16 path LDS layout ----------------
// floats: sqs[1024], Dist[16*1044]; then ushort: Ahi[16*264], Alo[16*264]
#define A2_STRIDE 264                                       /* bf16 elems per row */
#define LDS_BF16_BYTES ((1024 + 16 * D_STRIDE) * 4 + 2 * 16 * A2_STRIDE * 2) /* 87808 */

static __device__ __forceinline__ ushort_t bf16_rne(float x) {
  unsigned u = __builtin_bit_cast(unsigned, x);
  unsigned r = u + 0x7FFFu + ((u >> 16) & 1u);
  return (ushort_t)(r >> 16);
}
static __device__ __forceinline__ float bf16_up(ushort_t h) {
  return __builtin_bit_cast(float, (unsigned)h << 16);
}

struct U4x2 { uint4 a, b; };
static __device__ __forceinline__ v16bf mk16(uint4 a, uint4 b) {
  U4x2 t; t.a = a; t.b = b;
  return __builtin_bit_cast(v16bf, t);   // elems 0-7 from a, 8-15 from b
}

// ---------------------------------------------------------------------------
__global__ void sqnorm_kernel(const float* __restrict__ h, float* __restrict__ sq) {
  const int lane = threadIdx.x & 31;
  const int wv   = threadIdx.x >> 5;
  const int row  = blockIdx.x * 8 + wv;
  const float4* p = (const float4*)(h + (size_t)row * DD);
  float4 a = p[lane];
  float4 b = p[lane + 32];
  float acc = a.x * a.x + a.y * a.y + a.z * a.z + a.w * a.w +
              b.x * b.x + b.y * b.y + b.z * b.z + b.w * b.w;
#pragma unroll
  for (int off = 16; off > 0; off >>= 1) acc += __shfl_xor(acc, off, 32);
  if (lane == 0) sq[row] = acc;
}

// h -> (hi, lo) bf16 split, 4 elements per thread, RNE both times.
__global__ void split_kernel(const float* __restrict__ h,
                             ushort_t* __restrict__ hi, ushort_t* __restrict__ lo) {
  const size_t i4 = (size_t)blockIdx.x * 256 + threadIdx.x;
  float4 x = ((const float4*)h)[i4];
  ushort_t h0 = bf16_rne(x.x), h1 = bf16_rne(x.y), h2 = bf16_rne(x.z), h3 = bf16_rne(x.w);
  ushort_t l0 = bf16_rne(x.x - bf16_up(h0));
  ushort_t l1 = bf16_rne(x.y - bf16_up(h1));
  ushort_t l2 = bf16_rne(x.z - bf16_up(h2));
  ushort_t l3 = bf16_rne(x.w - bf16_up(h3));
  uint2 ph, pl;
  ph.x = (unsigned)h0 | ((unsigned)h1 << 16); ph.y = (unsigned)h2 | ((unsigned)h3 << 16);
  pl.x = (unsigned)l0 | ((unsigned)l1 << 16); pl.y = (unsigned)l2 | ((unsigned)l3 << 16);
  ((uint2*)hi)[i4] = ph;
  ((uint2*)lo)[i4] = pl;
}

// ---------------------------------------------------------------------------
// Shared epilogue: distances already in LDS Dist rows; 16 threads/row extract
// top-16 (ascending, ties -> smaller index), write dists/src/dst as floats.
static __device__ __forceinline__ void topk_epilogue(float* Dist, float* out,
                                                     int t, int b, int row_base) {
  const int m = t >> 4;
  const int g = t & 15;
  float* drow = Dist + m * D_STRIDE;
  const size_t gi   = (size_t)row_base + m;
  const size_t NTOT = (size_t)BB * LL * KK;
  const float  BIGF = 3.0e38f;
  for (int k = 0; k < KK; ++k) {
    float best = BIGF;
    int   bidx = LL;
    for (int j = g; j < LL; j += 16) {
      float v = drow[j];
      if (v < best) { best = v; bidx = j; }
    }
#pragma unroll
    for (int off = 8; off > 0; off >>= 1) {
      float ov = __shfl_xor(best, off, 16);
      int   oi = __shfl_xor(bidx, off, 16);
      if (ov < best || (ov == best && oi < bidx)) { best = ov; bidx = oi; }
    }
    if (g == k) {
      size_t o = gi * KK + k;
      out[o]            = best;
      out[NTOT + o]     = (float)gi;
      out[2 * NTOT + o] = (float)((b << 10) + bidx);
    }
    if (g == 0) drow[bidx] = BIGF;   // same-wave LDS ops are in-order
  }
}

// ---------------------------------------------------------------------------
// bf16x3 main kernel: Gram via v_wmma_f32_16x16x32_bf16.
__global__ void __launch_bounds__(256)
knn_bf16_kernel(const ushort_t* __restrict__ h_hi, const ushort_t* __restrict__ h_lo,
                const float* __restrict__ sqg, float* __restrict__ out) {
  extern __shared__ char smem[];
  float*    sqs  = (float*)smem;                         // 1024
  float*    Dist = sqs + 1024;                           // 16 x D_STRIDE
  ushort_t* Ahi  = (ushort_t*)(Dist + 16 * D_STRIDE);    // 16 x A2_STRIDE
  ushort_t* Alo  = Ahi + 16 * A2_STRIDE;

  const int t        = threadIdx.x;
  const int b        = blockIdx.x >> 6;
  const int ib       = blockIdx.x & 63;
  const int row_base = (b << 10) + (ib << 4);

  // Stage A block (16x256 bf16, hi+lo) and per-batch sq norms into LDS.
  for (int k2 = t; k2 < 16 * (DD / 2); k2 += 256) {       // copy as dwords
    int r = k2 >> 7, c = k2 & 127;
    ((unsigned*)(Ahi + r * A2_STRIDE))[c] =
        ((const unsigned*)(h_hi + (size_t)(row_base + r) * DD))[c];
    ((unsigned*)(Alo + r * A2_STRIDE))[c] =
        ((const unsigned*)(h_lo + (size_t)(row_base + r) * DD))[c];
  }
  for (int k2 = t; k2 < LL; k2 += 256) sqs[k2] = sqg[(b << 10) + k2];
  __syncthreads();

  const int wv   = t >> 5;
  const int lane = t & 31;
  const int half = lane >> 4;
  const int lm   = lane & 15;
  // A frag runs per lane: K in [d0+8*half, +8) and [d0+16+8*half, +8)
  const ushort_t* aH = Ahi + lm * A2_STRIDE + 8 * half;
  const ushort_t* aL = Alo + lm * A2_STRIDE + 8 * half;

  for (int jt = wv; jt < 64; jt += 8) {
    const int j0 = jt << 4;
    // B frag runs per lane: K in [d0+8*half, +8) and [d0+16+8*half, +8) of row (j0 + n)
    const ushort_t* bH = h_hi + ((size_t)(b << 10) + j0 + lm) * DD + 8 * half;
    const ushort_t* bL = h_lo + ((size_t)(b << 10) + j0 + lm) * DD + 8 * half;
    v8f c = {0.f, 0.f, 0.f, 0.f, 0.f, 0.f, 0.f, 0.f};
#pragma unroll
    for (int ch = 0; ch < 8; ++ch) {
      const int d0 = ch * 32;
      uint4 ah0 = *(const uint4*)(aH + d0);
      uint4 ah1 = *(const uint4*)(aH + d0 + 16);
      uint4 al0 = *(const uint4*)(aL + d0);
      uint4 al1 = *(const uint4*)(aL + d0 + 16);
      uint4 bh0 = *(const uint4*)(bH + d0);
      uint4 bh1 = *(const uint4*)(bH + d0 + 16);
      uint4 bl0 = *(const uint4*)(bL + d0);
      uint4 bl1 = *(const uint4*)(bL + d0 + 16);
      v16bf Ah = mk16(ah0, ah1), Al = mk16(al0, al1);
      v16bf Bh = mk16(bh0, bh1), Bl = mk16(bl0, bl1);
      c = __builtin_amdgcn_wmma_f32_16x16x32_bf16(false, Ah, false, Bh, (short)0, c, false, false);
      c = __builtin_amdgcn_wmma_f32_16x16x32_bf16(false, Ah, false, Bl, (short)0, c, false, false);
      c = __builtin_amdgcn_wmma_f32_16x16x32_bf16(false, Al, false, Bh, (short)0, c, false, false);
    }
    const float sqn = sqs[j0 + lm];
#pragma unroll
    for (int v = 0; v < 8; ++v) {
      const int m = (half << 3) + v;
      Dist[m * D_STRIDE + j0 + lm] = sqs[(ib << 4) + m] + sqn - 2.0f * c[v];
    }
  }
  __syncthreads();
  topk_epilogue(Dist, out, t, b, row_base);
}

// ---------------------------------------------------------------------------
// fp32 fallback (exact): Gram via v_wmma_f32_16x16x4_f32. Used if d_ws is too
// small for the bf16 split arrays.
__global__ void __launch_bounds__(256)
knn_f32_kernel(const float* __restrict__ h, const float* __restrict__ sqg,
               float* __restrict__ out) {
  extern __shared__ float fsmem[];
  float* As   = fsmem;
  float* sqs  = fsmem + SQ_OFF;
  float* Dist = fsmem + DIST_OFF;

  const int t        = threadIdx.x;
  const int b        = blockIdx.x >> 6;
  const int ib       = blockIdx.x & 63;
  const int row_base = (b << 10) + (ib << 4);

  for (int k2 = t; k2 < 16 * DD; k2 += 256) {
    int r = k2 >> 8, c = k2 & 255;
    As[r * A_STRIDE + c] = h[(size_t)(row_base + r) * DD + c];
  }
  for (int k2 = t; k2 < LL; k2 += 256) sqs[k2] = sqg[(b << 10) + k2];
  __syncthreads();

  const int wv   = t >> 5;
  const int lane = t & 31;
  const int half = lane >> 4;
  const int lm   = lane & 15;
  const int kk   = half << 1;
  const float* aptr = As + lm * A_STRIDE + kk;

  for (int jt = wv; jt < 64; jt += 8) {
    const int j0 = jt << 4;
    const float* bptr = h + ((size_t)(b << 10) + j0 + lm) * DD + kk;
    v8f c = {0.f, 0.f, 0.f, 0.f, 0.f, 0.f, 0.f, 0.f};
#pragma unroll 8
    for (int d0 = 0; d0 < DD; d0 += 4) {
      v2f av = *(const v2f*)(aptr + d0);
      v2f bv = *(const v2f*)(bptr + d0);
      c = __builtin_amdgcn_wmma_f32_16x16x4_f32(false, av, false, bv, (short)0, c, false, false);
    }
    const float sqn = sqs[j0 + lm];
#pragma unroll
    for (int v = 0; v < 8; ++v) {
      const int m = (half << 3) + v;
      Dist[m * D_STRIDE + j0 + lm] = sqs[(ib << 4) + m] + sqn - 2.0f * c[v];
    }
  }
  __syncthreads();
  topk_epilogue(Dist, out, t, b, row_base);
}

// ---------------------------------------------------------------------------
extern "C" void kernel_launch(void* const* d_in, const int* in_sizes, int n_in,
                              void* d_out, int out_size, void* d_ws, size_t ws_size,
                              hipStream_t stream) {
  (void)in_sizes; (void)n_in; (void)out_size;
  const float* h   = (const float*)d_in[0];
  float*       sq  = (float*)d_ws;                    // B*L fp32 norms (256 KB)
  const size_t N   = (size_t)BB * LL * DD;
  const size_t SQB = (size_t)BB * LL * sizeof(float);
  const size_t need = SQB + 2 * N * sizeof(ushort_t); // sq + hi + lo

  sqnorm_kernel<<<BB * LL / 8, 256, 0, stream>>>(h, sq);

  if (ws_size >= need) {
    // bf16x3 path: 2.7x fewer matrix instructions on the deep-K bf16 pipe.
    ushort_t* hi = (ushort_t*)((char*)d_ws + SQB);
    ushort_t* lo = hi + N;
    split_kernel<<<(int)(N / 1024), 256, 0, stream>>>(h, hi, lo);
    knn_bf16_kernel<<<BB * (LL / 16), 256, LDS_BF16_BYTES, stream>>>(hi, lo, sq,
                                                                     (float*)d_out);
  } else {
    knn_f32_kernel<<<BB * (LL / 16), 256, LDS_F32_FLOATS * sizeof(float), stream>>>(
        h, sq, (float*)d_out);
  }
}
